// MoELayer_23613730193627
// MI455X (gfx1250) — matrix operand order repaired and checked by reading, not compiled, not run on hardware
//
#include <hip/hip_runtime.h>
#include <hip/hip_bf16.h>

#define T_TOK   4096
#define D_MODEL 1024
#define D_FFN   4096
#define N_EXP   8
#define TOPK    2
#define NROWS   (T_TOK * TOPK)   // 8192 compacted token-expert rows

typedef __attribute__((ext_vector_type(16))) __bf16 v16bf;
typedef __attribute__((ext_vector_type(8)))  float  v8f;

// ---------- helpers ----------
__device__ __forceinline__ unsigned short f2bf(float f) {
    union { float f; unsigned u; } x; x.f = f;
    unsigned u = x.u + 0x7FFFu + ((x.u >> 16) & 1u);   // round-to-nearest-even
    return (unsigned short)(u >> 16);
}
__device__ __forceinline__ uint2 pack4(float4 v) {
    uint2 r;
    r.x = (unsigned)f2bf(v.x) | ((unsigned)f2bf(v.y) << 16);
    r.y = (unsigned)f2bf(v.z) | ((unsigned)f2bf(v.w) << 16);
    return r;
}
union FragBF { uint4 q[2]; v16bf v; };
__device__ __forceinline__ v16bf load_frag(const unsigned short* lds, int e0, int e1) {
    FragBF f;
    f.q[0] = *(const uint4*)(lds + e0);
    f.q[1] = *(const uint4*)(lds + e1);
    return f.v;
}
__device__ __forceinline__ void wait_async0() {
#if __has_builtin(__builtin_amdgcn_s_wait_asynccnt)
    __builtin_amdgcn_s_wait_asynccnt(0);
#else
    asm volatile("s_wait_asynccnt 0x0" ::: "memory");
#endif
}
// 32 bytes memory -> LDS, no VGPR round trip (ASYNCcnt-tracked, GV mode)
__device__ __forceinline__ void async_copy_b256(unsigned lds_addr, unsigned long long gaddr) {
    asm volatile(
        "global_load_async_to_lds_b128 %0, %1, off\n\t"
        "global_load_async_to_lds_b128 %0, %1, off offset:16"
        :: "v"(lds_addr), "v"(gaddr) : "memory");
}

// ---------- 0) init: zero out + stats ----------
__global__ void init_k(float* out, int n, int* counts, float* probsum) {
    int i = blockIdx.x * blockDim.x + threadIdx.x;
    if (i < n) out[i] = 0.0f;
    if (blockIdx.x == 0 && threadIdx.x < N_EXP) {
        counts[threadIdx.x]  = 0;
        probsum[threadIdx.x] = 0.0f;
    }
}

// ---------- 1) router: one wave32 per token ----------
__global__ __launch_bounds__(256) void router_k(const float* __restrict__ x,
                                                const float* __restrict__ wr,
                                                int* topk_idx, float* topk_p,
                                                int* counts, float* probsum) {
    int wid  = threadIdx.x >> 5;
    int lane = threadIdx.x & 31;
    int t    = blockIdx.x * 8 + wid;
    if (t >= T_TOK) return;

    float acc[N_EXP] = {0.f,0.f,0.f,0.f,0.f,0.f,0.f,0.f};
    const float4* w4 = (const float4*)wr;           // row d -> float4 pair
    for (int d = lane; d < D_MODEL; d += 32) {
        float  xv = x[(size_t)t * D_MODEL + d];
        float4 a  = w4[d * 2 + 0];
        float4 b  = w4[d * 2 + 1];
        acc[0] += xv * a.x; acc[1] += xv * a.y; acc[2] += xv * a.z; acc[3] += xv * a.w;
        acc[4] += xv * b.x; acc[5] += xv * b.y; acc[6] += xv * b.z; acc[7] += xv * b.w;
    }
    #pragma unroll
    for (int e = 0; e < N_EXP; ++e)
        #pragma unroll
        for (int o = 16; o > 0; o >>= 1)
            acc[e] += __shfl_down(acc[e], o, 32);

    if (lane == 0) {
        float m = acc[0];
        #pragma unroll
        for (int e = 1; e < N_EXP; ++e) m = fmaxf(m, acc[e]);
        float p[N_EXP], s = 0.f;
        #pragma unroll
        for (int e = 0; e < N_EXP; ++e) { p[e] = __expf(acc[e] - m); s += p[e]; }
        float inv = 1.0f / s;
        int e1 = 0; float b1 = -1.0f;
        #pragma unroll
        for (int e = 0; e < N_EXP; ++e) {
            float pe = p[e] * inv; p[e] = pe;
            atomicAdd(&probsum[e], pe);
            if (pe > b1) { b1 = pe; e1 = e; }       // ties -> lowest index
        }
        int e2 = 0; float b2 = -1.0f;
        #pragma unroll
        for (int e = 0; e < N_EXP; ++e)
            if (e != e1 && p[e] > b2) { b2 = p[e]; e2 = e; }
        float sn = b1 + b2;
        topk_idx[t * 2 + 0] = e1;  topk_idx[t * 2 + 1] = e2;
        topk_p  [t * 2 + 0] = b1 / sn;  topk_p[t * 2 + 1] = b2 / sn;
        atomicAdd(&counts[e1], 1);
        atomicAdd(&counts[e2], 1);
    }
}

// ---------- 2) build compacted per-expert row lists (single block) ----------
__global__ void build_lists_k(const int* __restrict__ topk_idx,
                              const float* __restrict__ topk_p,
                              const int* __restrict__ counts,
                              int* offs, int* cursor,
                              int* row_token, float* row_wt) {
    if (threadIdx.x == 0) {
        int s = 0;
        for (int e = 0; e < N_EXP; ++e) { offs[e] = s; cursor[e] = s; s += counts[e]; }
    }
    __syncthreads();
    for (int i = threadIdx.x; i < NROWS; i += blockDim.x) {
        int e   = topk_idx[i];
        int pos = atomicAdd(&cursor[e], 1);
        row_token[pos] = i >> 1;
        row_wt[pos]    = topk_p[i];
    }
}

// ---------- 3) GEMM1: a = silu(X W1) * (X W3), routed rows, bf16 WMMA ----------
__global__ __launch_bounds__(256) void gemm1_k(const float* __restrict__ x,
                                               const float* __restrict__ w1,
                                               const float* __restrict__ w3,
                                               const int* __restrict__ rows,
                                               const int* __restrict__ counts,
                                               const int* __restrict__ offs,
                                               unsigned short* __restrict__ act) {
    const int bx = blockIdx.x;
    const int e  = bx >> 10;            // 32 n-tiles * 32 m-tiles per expert
    const int nt = (bx >> 5) & 31;
    const int mt = bx & 31;
    const int cnt = counts[e], off = offs[e];
    if (mt * 128 >= cnt) return;

    __shared__ unsigned short sA [2][128 * 32];   // [m][k]
    __shared__ unsigned short sB1[2][128 * 32];   // transposed [n][k]
    __shared__ unsigned short sB3[2][128 * 32];

    const int tid  = threadIdx.x;
    const int wid  = tid >> 5, lane = tid & 31;
    const int half = lane >> 4, lr = lane & 15;
    const int wm   = wid & 3,  wn = wid >> 2;     // 4x2 wave grid over 128x128

    v8f ah[2][4] = {}, ag[2][4] = {};
    const int KS = D_MODEL / 32;                  // 32 k-steps

    auto load_stage = [&](int buf, int kk) {
        const int k0 = kk * 32;
        {   // A: gathered token rows, fp32 -> bf16
            int m = tid >> 1, seg = tid & 1;
            int gr = mt * 128 + m;
            int t  = rows[off + (gr < cnt ? gr : 0)];
            const float4* src = (const float4*)(x + (size_t)t * D_MODEL + k0 + seg * 16);
            unsigned short* dst = &sA[buf][m * 32 + seg * 16];
            #pragma unroll
            for (int i = 0; i < 4; ++i) *(uint2*)(dst + i * 4) = pack4(src[i]);
        }
        {   // B1/B3: fp32 -> bf16, transposed into LDS
            int k = tid & 31, nc = tid >> 5;
            const float* s1 = w1 + ((size_t)e * D_MODEL + k0 + k) * D_FFN + nt * 128 + nc * 16;
            const float* s3 = w3 + ((size_t)e * D_MODEL + k0 + k) * D_FFN + nt * 128 + nc * 16;
            if (kk + 1 < KS) {
                __builtin_prefetch(s1 + (size_t)32 * D_FFN, 0, 0);   // global_prefetch_b8
                __builtin_prefetch(s3 + (size_t)32 * D_FFN, 0, 0);
            }
            #pragma unroll
            for (int i = 0; i < 4; ++i) {
                float4 a = ((const float4*)s1)[i];
                float4 b = ((const float4*)s3)[i];
                int nb = nc * 16 + i * 4;
                sB1[buf][(nb + 0) * 32 + k] = f2bf(a.x); sB1[buf][(nb + 1) * 32 + k] = f2bf(a.y);
                sB1[buf][(nb + 2) * 32 + k] = f2bf(a.z); sB1[buf][(nb + 3) * 32 + k] = f2bf(a.w);
                sB3[buf][(nb + 0) * 32 + k] = f2bf(b.x); sB3[buf][(nb + 1) * 32 + k] = f2bf(b.y);
                sB3[buf][(nb + 2) * 32 + k] = f2bf(b.z); sB3[buf][(nb + 3) * 32 + k] = f2bf(b.w);
            }
        }
    };

    load_stage(0, 0);
    __syncthreads();
    for (int kk = 0; kk < KS; ++kk) {
        const int cur = kk & 1;
        if (kk + 1 < KS) load_stage(cur ^ 1, kk + 1);
        v16bf af[2];
        #pragma unroll
        for (int mi = 0; mi < 2; ++mi) {
            int base = (wm * 32 + mi * 16 + lr) * 32;
            af[mi] = load_frag(&sA[cur][0], base + half * 8, base + 16 + half * 8);
        }
        #pragma unroll
        for (int ni = 0; ni < 4; ++ni) {
            int nb = (wn * 64 + ni * 16 + lr) * 32 + half * 8;
            v16bf b1 = load_frag(&sB1[cur][0], nb, nb + 16);
            v16bf b3 = load_frag(&sB3[cur][0], nb, nb + 16);
            #pragma unroll
            for (int mi = 0; mi < 2; ++mi) {
                ah[mi][ni] = __builtin_amdgcn_wmma_f32_16x16x32_bf16(
                    false, af[mi], false, b1, (short)0, ah[mi][ni], false, false);
                ag[mi][ni] = __builtin_amdgcn_wmma_f32_16x16x32_bf16(
                    false, af[mi], false, b3, (short)0, ag[mi][ni], false, false);
            }
        }
        __syncthreads();
    }

    // fused SiLU(h)*g epilogue -> bf16 activation rows
    #pragma unroll
    for (int mi = 0; mi < 2; ++mi)
        #pragma unroll
        for (int v = 0; v < 8; ++v) {
            int ml = wm * 32 + mi * 16 + half * 8 + v;
            int gr = mt * 128 + ml;
            if (gr < cnt) {
                #pragma unroll
                for (int ni = 0; ni < 4; ++ni) {
                    int nl = wn * 64 + ni * 16 + lr;
                    float h = ah[mi][ni][v], g = ag[mi][ni][v];
                    float a = (h / (1.0f + __expf(-h))) * g;
                    act[(size_t)(off + gr) * D_FFN + nt * 128 + nl] = f2bf(a);
                }
            }
        }
}

// ---------- 4) GEMM2: y = a W2, scale by routing weight, scatter-add ----------
__global__ __launch_bounds__(256) void gemm2_k(const unsigned short* __restrict__ act,
                                               const float* __restrict__ w2,
                                               const int* __restrict__ row_token,
                                               const float* __restrict__ row_wt,
                                               const int* __restrict__ counts,
                                               const int* __restrict__ offs,
                                               float* __restrict__ out) {
    const int bx = blockIdx.x;
    const int e  = bx >> 8;             // 8 n-tiles * 32 m-tiles per expert
    const int nt = (bx >> 5) & 7;
    const int mt = bx & 31;
    const int cnt = counts[e], off = offs[e];
    if (mt * 128 >= cnt) return;

    __shared__ unsigned short sA[2][128 * 32];    // [m][k], bf16 already
    __shared__ unsigned short sB[2][128 * 32];    // transposed [n][k]

    const int tid  = threadIdx.x;
    const int wid  = tid >> 5, lane = tid & 31;
    const int half = lane >> 4, lr = lane & 15;
    const int wm   = wid & 3,  wn = wid >> 2;

    v8f acc[2][4] = {};
    const int KS = D_FFN / 32;                    // 128 k-steps

    auto load_stage = [&](int buf, int kk) {
        const int k0 = kk * 32;
        {   // A: contiguous compacted bf16 rows, async DMA global->LDS (no VGPR hop)
            int m = tid >> 1, seg = tid & 1;
            size_t ar = (size_t)off + mt * 128 + m;
            if (ar >= NROWS) ar = NROWS - 1;      // clamp (discarded by cnt guard)
            unsigned long long gaddr =
                (unsigned long long)(act + ar * D_FFN + k0 + seg * 16);
            unsigned lds_a =
                (unsigned)(uintptr_t)(&sA[buf][m * 32 + seg * 16]);  // low 32b = LDS offset
            async_copy_b256(lds_a, gaddr);        // 2x global_load_async_to_lds_b128
        }
        {   // B: w2 fp32 -> bf16, transposed
            int k = tid & 31, nc = tid >> 5;
            const float* s2 = w2 + ((size_t)e * D_FFN + k0 + k) * D_MODEL + nt * 128 + nc * 16;
            if (kk + 1 < KS) __builtin_prefetch(s2 + (size_t)32 * D_MODEL, 0, 0);
            #pragma unroll
            for (int i = 0; i < 4; ++i) {
                float4 a = ((const float4*)s2)[i];
                int nb = nc * 16 + i * 4;
                sB[buf][(nb + 0) * 32 + k] = f2bf(a.x); sB[buf][(nb + 1) * 32 + k] = f2bf(a.y);
                sB[buf][(nb + 2) * 32 + k] = f2bf(a.z); sB[buf][(nb + 3) * 32 + k] = f2bf(a.w);
            }
        }
    };

    load_stage(0, 0);
    wait_async0();
    __syncthreads();
    for (int kk = 0; kk < KS; ++kk) {
        const int cur = kk & 1;
        if (kk + 1 < KS) load_stage(cur ^ 1, kk + 1);   // async A overlaps WMMA below
        v16bf af[2];
        #pragma unroll
        for (int mi = 0; mi < 2; ++mi) {
            int base = (wm * 32 + mi * 16 + lr) * 32;
            af[mi] = load_frag(&sA[cur][0], base + half * 8, base + 16 + half * 8);
        }
        #pragma unroll
        for (int ni = 0; ni < 4; ++ni) {
            int nb = (wn * 64 + ni * 16 + lr) * 32 + half * 8;
            v16bf bf = load_frag(&sB[cur][0], nb, nb + 16);
            #pragma unroll
            for (int mi = 0; mi < 2; ++mi)
                acc[mi][ni] = __builtin_amdgcn_wmma_f32_16x16x32_bf16(
                    false, af[mi], false, bf, (short)0, acc[mi][ni], false, false);
        }
        wait_async0();                                   // fence this stage's async fills
        __syncthreads();
    }

    // weighted scatter-add into out (each token accumulates 2 experts)
    #pragma unroll
    for (int mi = 0; mi < 2; ++mi)
        #pragma unroll
        for (int v = 0; v < 8; ++v) {
            int ml = wm * 32 + mi * 16 + half * 8 + v;
            int gr = mt * 128 + ml;
            if (gr < cnt) {
                int r  = off + gr;
                int t  = row_token[r];
                float w = row_wt[r];
                #pragma unroll
                for (int ni = 0; ni < 4; ++ni) {
                    int nl = wn * 64 + ni * 16 + lr;
                    atomicAdd(&out[(size_t)t * D_MODEL + nt * 128 + nl], acc[mi][ni][v] * w);
                }
            }
        }
}

// ---------- 5) aux loss ----------
__global__ void aux_k(const int* counts, const float* probsum, float* out_aux) {
    if (blockIdx.x == 0 && threadIdx.x == 0) {
        float s = 0.f;
        for (int e = 0; e < N_EXP; ++e) s += (float)counts[e] * probsum[e];
        *out_aux = (float)N_EXP * s / ((float)T_TOK * (float)T_TOK);
    }
}

// ---------- launch ----------
extern "C" void kernel_launch(void* const* d_in, const int* in_sizes, int n_in,
                              void* d_out, int out_size, void* d_ws, size_t ws_size,
                              hipStream_t stream) {
    const float* x  = (const float*)d_in[0];
    const float* wr = (const float*)d_in[1];
    const float* w1 = (const float*)d_in[2];
    const float* w2 = (const float*)d_in[3];
    const float* w3 = (const float*)d_in[4];
    float* out = (float*)d_out;

    char* base = (char*)d_ws;
    const size_t ACT_BYTES = (size_t)NROWS * D_FFN * 2;   // 64 MB bf16 activations
    unsigned short* act = (unsigned short*)base;
    int*   counts    = (int*)(base + ACT_BYTES);
    int*   offs      = counts + 8;
    int*   cursor    = counts + 16;
    float* probsum   = (float*)(counts + 24);
    int*   topk_idx  = (int*)(base + ACT_BYTES + 128);
    float* topk_p    = (float*)((char*)topk_idx + NROWS * 4);
    int*   row_token = (int*)((char*)topk_p + NROWS * 4);
    float* row_wt    = (float*)((char*)row_token + NROWS * 4);

    const int n_out = T_TOK * D_MODEL + 1;
    init_k<<<(n_out + 255) / 256, 256, 0, stream>>>(out, n_out, counts, probsum);
    router_k<<<T_TOK / 8, 256, 0, stream>>>(x, wr, topk_idx, topk_p, counts, probsum);
    build_lists_k<<<1, 256, 0, stream>>>(topk_idx, topk_p, counts, offs, cursor,
                                         row_token, row_wt);
    gemm1_k<<<N_EXP * 32 * 32, 256, 0, stream>>>(x, w1, w3, row_token, counts, offs, act);
    gemm2_k<<<N_EXP * 8 * 32, 256, 0, stream>>>(act, w2, row_token, row_wt, counts, offs, out);
    aux_k<<<1, 32, 0, stream>>>(counts, probsum, out + (size_t)T_TOK * D_MODEL);
}
